// SimilarityModel_72215580115002
// MI455X (gfx1250) — matrix operand order repaired and checked by reading, not compiled, not run on hardware
//
#include <hip/hip_runtime.h>
#include <cstdint>
#include <cstddef>

// MI455X / gfx1250, wave32. Split-bf16 (hi+lo) fp32-emulated GEMMs via
// v_wmma_f32_16x16x32_bf16 (3 WMMAs per K-chunk: hi*hi + hi*lo + lo*hi).
// B operand strip staged in LDS once per block (8x reuse across waves).

#define D_DIM 128
#define BK    136   // padded LDS row stride (u16) for B tile: 68 dwords

typedef __attribute__((ext_vector_type(16))) __bf16 v16bf;
typedef __attribute__((ext_vector_type(8)))  __bf16 v8bf;
typedef __attribute__((ext_vector_type(8)))  float  v8f;
typedef unsigned short u16;

__device__ __forceinline__ u16 f2bf(float f) {
  unsigned u = __float_as_uint(f);
  u += 0x7FFFu + ((u >> 16) & 1u);   // round-to-nearest-even
  return (u16)(u >> 16);
}
__device__ __forceinline__ float bf2f(u16 h) {
  return __uint_as_float(((unsigned)h) << 16);
}

// ---------------- split: fp32 -> bf16 hi + bf16 lo ----------------
__global__ void split_kernel(const float* __restrict__ x, u16* __restrict__ hi,
                             u16* __restrict__ lo, int n) {
  int i = blockIdx.x * 256 + threadIdx.x;
  if (i >= n) return;
  float v = x[i];
  u16 h = f2bf(v);
  hi[i] = h;
  lo[i] = f2bf(v - bf2f(h));
}

// ---------- row-normalize (D=128) + split, one row per block ----------
__global__ void normsplit_kernel(const float* __restrict__ x, u16* __restrict__ hi,
                                 u16* __restrict__ lo) {
  __shared__ float red[128];
  const int t = threadIdx.x;                 // 0..127
  const size_t base = (size_t)blockIdx.x * D_DIM;
  float v = x[base + t];
  red[t] = v * v;
  __syncthreads();
  for (int s = 64; s > 0; s >>= 1) {
    if (t < s) red[t] += red[t + s];
    __syncthreads();
  }
  float rn = 1.0f / fmaxf(sqrtf(red[0]), 1e-8f);
  float nv = v * rn;
  u16 h = f2bf(nv);
  hi[base + t] = h;
  lo[base + t] = f2bf(nv - bf2f(h));
}

// --------- per-row softmax stats over M columns: max and 1/sumexp ---------
__global__ void stats_kernel(const float* __restrict__ S, float* __restrict__ rowmax,
                             float* __restrict__ rowinv, int M) {
  __shared__ float red[256];
  const int t = threadIdx.x;
  const float* s = S + (size_t)blockIdx.x * M;
  float m = -3.4e38f;
  for (int j = t; j < M; j += 256) m = fmaxf(m, s[j]);
  red[t] = m;
  __syncthreads();
  for (int k = 128; k > 0; k >>= 1) { if (t < k) red[t] = fmaxf(red[t], red[t + k]); __syncthreads(); }
  m = red[0];
  __syncthreads();
  float a = 0.0f;
  for (int j = t; j < M; j += 256) a += __expf(s[j] - m);
  red[t] = a;
  __syncthreads();
  for (int k = 128; k > 0; k >>= 1) { if (t < k) red[t] += red[t + k]; __syncthreads(); }
  if (t == 0) { rowmax[blockIdx.x] = m; rowinv[blockIdx.x] = 1.0f / red[0]; }
}

// ---------------- WMMA fragment loaders ----------------
// A 16x32 bf16 frag (global, row stride 128):
//   lane m=lane&15; elems 0..7 = K[ka..ka+7], 8..15 = K[ka+16..ka+23]
__device__ __forceinline__ v16bf load_frag_a(const u16* __restrict__ p) {
  v16bf r;
  ((v8bf*)&r)[0] = *(const v8bf*)(p);
  ((v8bf*)&r)[1] = *(const v8bf*)(p + 16);
  return r;
}
// B 32x16 bf16 frag (LDS, padded stride): elems 0..7 = K[kb..kb+7], 8..15 = K[kb+16..kb+23]
__device__ __forceinline__ v16bf load_frag_b(const u16* p) {
  v16bf r;
  ((v8bf*)&r)[0] = *(const v8bf*)(p);
  ((v8bf*)&r)[1] = *(const v8bf*)(p + 16);
  return r;
}

// ---------------- split-bf16 GEMM: C[rows x Mcols] = A * B^T ----------------
// Block = 256 threads = 8 waves. Block tile 128 rows x 64 cols; each wave 16x64.
// B strip (64 rows of B = 64 output cols, full K=128, hi+lo) staged in LDS.
// MODE 0: out = acc (raw scores)
// MODE 1: out = relu(acc) * exp(out_prev - rowmax[row]) * rowinv[row]   (finalize)
// MODE 2: out = acc + bias[col]                                          (q/k proj)
template <int MODE>
__global__ __launch_bounds__(256) void gemm_split(
    const u16* __restrict__ Ahi, const u16* __restrict__ Alo,
    const u16* __restrict__ Bhi, const u16* __restrict__ Blo,
    float* __restrict__ out,
    const float* __restrict__ rowmax, const float* __restrict__ rowinv,
    const float* __restrict__ bias, int Mcols) {
  __shared__ __align__(16) u16 lBH[64 * BK];
  __shared__ __align__(16) u16 lBL[64 * BK];

  const int lane = threadIdx.x & 31;
  const int wid  = threadIdx.x >> 5;
  const int ln   = lane & 15;
  const int lhi  = lane >> 4;
  const int rowBase = blockIdx.y * 128 + wid * 16;
  const int colBase = blockIdx.x * 64;

  // ---- stage B strip: 64 rows x 128 K, contiguous in global, coalesced 16B chunks
  {
    const u16* gH = Bhi + (size_t)colBase * D_DIM;
    const u16* gL = Blo + (size_t)colBase * D_DIM;
    for (int c = threadIdx.x; c < 1024; c += 256) {   // 1024 chunks of 8 u16
      const int j   = c * 8;
      const int row = j >> 7;          // 0..63
      const int col = j & 127;         // 0..120, chunks never cross a row
      *(v8bf*)&lBH[row * BK + col] = *(const v8bf*)(gH + j);
      *(v8bf*)&lBL[row * BK + col] = *(const v8bf*)(gL + j);
    }
  }
  __syncthreads();

  v8f acc[4] = {};

  const u16* paH = Ahi + (size_t)(rowBase + ln) * D_DIM;
  const u16* paL = Alo + (size_t)(rowBase + ln) * D_DIM;

#pragma unroll
  for (int kc = 0; kc < 4; ++kc) {
    const int ka = kc * 32 + lhi * 8;    // A: 8 @ ka, 8 @ ka+16
    const int kb = kc * 32 + lhi * 8;    // B: 8 @ kb, 8 @ kb+16
    v16bf aH = load_frag_a(paH + ka);
    v16bf aL = load_frag_a(paL + ka);
#pragma unroll
    for (int ct = 0; ct < 4; ++ct) {
      const int boff = (ct * 16 + ln) * BK + kb;
      v16bf bH = load_frag_b(&lBH[boff]);
      v16bf bL = load_frag_b(&lBL[boff]);
      acc[ct] = __builtin_amdgcn_wmma_f32_16x16x32_bf16(false, aH, false, bL, (short)0, acc[ct], false, false);
      acc[ct] = __builtin_amdgcn_wmma_f32_16x16x32_bf16(false, aL, false, bH, (short)0, acc[ct], false, false);
      acc[ct] = __builtin_amdgcn_wmma_f32_16x16x32_bf16(false, aH, false, bH, (short)0, acc[ct], false, false);
    }
  }

  // C/D layout: vgpr r -> row = rowBase + 8*lhi + r, col = colBase + 16*ct + ln
  if (MODE == 0) {
#pragma unroll
    for (int ct = 0; ct < 4; ++ct) {
      const int col = colBase + ct * 16 + ln;
#pragma unroll
      for (int r = 0; r < 8; ++r) {
        const int row = rowBase + lhi * 8 + r;
        out[(size_t)row * Mcols + col] = acc[ct][r];
      }
    }
  } else if (MODE == 2) {
#pragma unroll
    for (int ct = 0; ct < 4; ++ct) {
      const int col = colBase + ct * 16 + ln;
      const float b = bias[col];
#pragma unroll
      for (int r = 0; r < 8; ++r) {
        const int row = rowBase + lhi * 8 + r;
        out[(size_t)row * Mcols + col] = acc[ct][r] + b;
      }
    }
  } else {
    float rm[8], ri[8];
#pragma unroll
    for (int r = 0; r < 8; ++r) {
      const int row = rowBase + lhi * 8 + r;
      rm[r] = rowmax[row];
      ri[r] = rowinv[row];
    }
#pragma unroll
    for (int ct = 0; ct < 4; ++ct) {
      const int col = colBase + ct * 16 + ln;
#pragma unroll
      for (int r = 0; r < 8; ++r) {
        const int row = rowBase + lhi * 8 + r;
        const size_t idx = (size_t)row * Mcols + col;
        const float w = __expf(out[idx] - rm[r]) * ri[r];
        out[idx] = fmaxf(acc[ct][r], 0.0f) * w;
      }
    }
  }
}

extern "C" void kernel_launch(void* const* d_in, const int* in_sizes, int n_in,
                              void* d_out, int out_size, void* d_ws, size_t ws_size,
                              hipStream_t stream) {
  const float* sub  = (const float*)d_in[0];
  const float* data = (const float*)d_in[1];
  const float* Wq   = (const float*)d_in[2];
  const float* bq   = (const float*)d_in[3];
  const float* Wk   = (const float*)d_in[4];
  const float* bk   = (const float*)d_in[5];
  float* out = (float*)d_out;

  const int N = in_sizes[0] / D_DIM;   // 4096
  const int M = in_sizes[1] / D_DIM;   // 8192
  const int ND = N * D_DIM, MD = M * D_DIM, DD = D_DIM * D_DIM;

  // workspace carve (256B aligned), total ~25 MB
  char* p = (char*)d_ws;
  auto carve = [&](size_t bytes) -> char* {
    char* q = p;
    p += (bytes + 255) & ~(size_t)255;
    return q;
  };
  u16* SubHi = (u16*)carve((size_t)ND * 2);  u16* SubLo = (u16*)carve((size_t)ND * 2);
  u16* DatHi = (u16*)carve((size_t)MD * 2);  u16* DatLo = (u16*)carve((size_t)MD * 2);
  u16* WqHi  = (u16*)carve((size_t)DD * 2);  u16* WqLo  = (u16*)carve((size_t)DD * 2);
  u16* WkHi  = (u16*)carve((size_t)DD * 2);  u16* WkLo  = (u16*)carve((size_t)DD * 2);
  float* qf  = (float*)carve((size_t)ND * 4);
  float* kf  = (float*)carve((size_t)MD * 4);
  u16* Qhi = (u16*)carve((size_t)ND * 2);    u16* Qlo = (u16*)carve((size_t)ND * 2);
  u16* Khi = (u16*)carve((size_t)MD * 2);    u16* Klo = (u16*)carve((size_t)MD * 2);
  u16* SNhi = (u16*)carve((size_t)ND * 2);   u16* SNlo = (u16*)carve((size_t)ND * 2);
  u16* DNhi = (u16*)carve((size_t)MD * 2);   u16* DNlo = (u16*)carve((size_t)MD * 2);
  float* rowmax = (float*)carve((size_t)N * 4);
  float* rowinv = (float*)carve((size_t)N * 4);

  // 1) split raw operands to bf16 hi/lo
  split_kernel<<<(ND + 255) / 256, 256, 0, stream>>>(sub,  SubHi, SubLo, ND);
  split_kernel<<<(MD + 255) / 256, 256, 0, stream>>>(data, DatHi, DatLo, MD);
  split_kernel<<<(DD + 255) / 256, 256, 0, stream>>>(Wq,   WqHi,  WqLo,  DD);
  split_kernel<<<(DD + 255) / 256, 256, 0, stream>>>(Wk,   WkHi,  WkLo,  DD);

  // 2) projections: q = sub*Wq^T + bq ; k = data*Wk^T + bk   (WMMA, bias epilogue)
  gemm_split<2><<<dim3(D_DIM / 64, N / 128), 256, 0, stream>>>(
      SubHi, SubLo, WqHi, WqLo, qf, nullptr, nullptr, bq, D_DIM);
  gemm_split<2><<<dim3(D_DIM / 64, M / 128), 256, 0, stream>>>(
      DatHi, DatLo, WkHi, WkLo, kf, nullptr, nullptr, bk, D_DIM);

  // 3) split q/k for the big GEMM; normalize+split sub/data for cosine
  split_kernel<<<(ND + 255) / 256, 256, 0, stream>>>(qf, Qhi, Qlo, ND);
  split_kernel<<<(MD + 255) / 256, 256, 0, stream>>>(kf, Khi, Klo, MD);
  normsplit_kernel<<<N, 128, 0, stream>>>(sub,  SNhi, SNlo);
  normsplit_kernel<<<M, 128, 0, stream>>>(data, DNhi, DNlo);

  // 4) scores = q @ k^T   -> stored in d_out (exactly N*M fp32)
  gemm_split<0><<<dim3(M / 64, N / 128), 256, 0, stream>>>(
      Qhi, Qlo, Khi, Klo, out, nullptr, nullptr, nullptr, M);

  // 5) per-row softmax stats
  stats_kernel<<<N, 256, 0, stream>>>(out, rowmax, rowinv, M);

  // 6) cos = relu(sub_n @ data_n^T); out = cos * softmax(scores) in place
  gemm_split<1><<<dim3(M / 64, N / 128), 256, 0, stream>>>(
      SNhi, SNlo, DNhi, DNlo, out, rowmax, rowinv, nullptr, M);
}
